// RNNRegressor_60816736911510
// MI455X (gfx1250) — matrix operand order repaired and checked by reading, not compiled, not run on hardware
//
#include <hip/hip_runtime.h>
#include <hip/hip_bf16.h>
#include <math.h>

// ---------------------------------------------------------------- types
typedef __bf16 bf16_t;
typedef __attribute__((ext_vector_type(16))) __bf16 v16bf;
typedef __attribute__((ext_vector_type(8)))  __bf16 bf16x8;
typedef __attribute__((ext_vector_type(8)))  float  v8f;
typedef __attribute__((ext_vector_type(4)))  float  f32x4;

#define B_  64
#define T_  1024
#define I_  256
#define H_  512
#define O_  64
#define HPAD 520   // LDS row stride in bf16 elems: 1040B -> 16B-aligned, 16-byte bank skew per row

// hardware tanh on gfx1250 (v_tanh_f32) with safe fallback
__device__ __forceinline__ float fast_tanh(float x) {
#if defined(__has_builtin)
#if __has_builtin(__builtin_amdgcn_tanhf)
    return __builtin_amdgcn_tanhf(x);
#elif __has_builtin(__builtin_amdgcn_tanh_f32)
    return __builtin_amdgcn_tanh_f32(x);
#else
    return tanhf(x);
#endif
#else
    return tanhf(x);
#endif
}

__device__ __forceinline__ v16bf pack16(bf16x8 lo, bf16x8 hi) {
    v16bf r;
#pragma unroll
    for (int i = 0; i < 8; ++i) { r[i] = lo[i]; r[i + 8] = hi[i]; }
    return r;
}

// load 8 consecutive f32 and round to bf16
__device__ __forceinline__ bf16x8 cvt8(const float* __restrict__ p) {
    f32x4 a = *(const f32x4*)p;
    f32x4 b = *(const f32x4*)(p + 4);
    bf16x8 r;
#pragma unroll
    for (int i = 0; i < 4; ++i) { r[i] = (__bf16)a[i]; r[i + 4] = (__bf16)b[i]; }
    return r;
}

// ---------------------------------------------------------------- helpers
__global__ void cvt_f32_to_bf16(const float* __restrict__ src, bf16_t* __restrict__ dst, int n) {
    int i = blockIdx.x * 256 + threadIdx.x;
    if (i < n) dst[i] = (bf16_t)src[i];
}

__global__ void bias_sum_k(const float* __restrict__ a, const float* __restrict__ b,
                           float* __restrict__ o) {
    int i = threadIdx.x;           // 512 threads
    o[i] = a[i] + b[i];
}

// ---------------------------------------------------------------- batched GEMM
// C[m][n] = sum_k A[m][k] * W[n][k] + bias[n]     (A: M=65536 rows)
// block = 256 thr (8 waves); wave -> 16 rows x 64 cols (4 WMMA tiles); grid = (M/128, N/64)
// TM: store xp time-major  C[(t*B + b)*H + n] with m = b*T + t
template <bool AF32, bool TM>
__global__ __launch_bounds__(256) void proj_gemm(
    const void* __restrict__ Ap, const bf16_t* __restrict__ W,
    const float* __restrict__ bias, float* __restrict__ C,
    int K, int Ncols)
{
    const int lane = threadIdx.x & 31;
    const int wave = threadIdx.x >> 5;
    const int half = lane >> 4;
    const int nl   = lane & 15;
    const int m_base = (blockIdx.x * 8 + wave) * 16;
    const int n0 = blockIdx.y * 64;

    v8f acc[4];
#pragma unroll
    for (int tt = 0; tt < 4; ++tt) acc[tt] = (v8f)0.f;

    const int ksteps = K >> 5;
    for (int ks = 0; ks < ksteps; ++ks) {
        const int arow = m_base + nl;
        const int kb = ks * 32 + half * 8;
        bf16x8 a0, a1;
        if (AF32) {
            const float* A = (const float*)Ap + (size_t)arow * K;
            a0 = cvt8(A + kb);
            a1 = cvt8(A + kb + 16);
        } else {
            const bf16_t* A = (const bf16_t*)Ap + (size_t)arow * K;
            a0 = *(const bf16x8*)(A + kb);
            a1 = *(const bf16x8*)(A + kb + 16);
        }
        v16bf af = pack16(a0, a1);
#pragma unroll
        for (int tt = 0; tt < 4; ++tt) {
            const int n = n0 + tt * 16 + nl;
            const bf16_t* wp = W + (size_t)n * K + ks * 32 + half * 8;
            v16bf bfm = pack16(*(const bf16x8*)wp, *(const bf16x8*)(wp + 16));
            acc[tt] = __builtin_amdgcn_wmma_f32_16x16x32_bf16(
                false, af, false, bfm, (short)0, acc[tt], false, false);
        }
    }

#pragma unroll
    for (int tt = 0; tt < 4; ++tt) {
        const int n = n0 + tt * 16 + nl;
        const float bv = bias[n];
#pragma unroll
        for (int r = 0; r < 8; ++r) {
            const int m = m_base + r + half * 8;
            const float v = acc[tt][r] + bv;
            size_t off;
            if (TM)  off = ((size_t)(m & (T_ - 1)) * B_ + (m >> 10)) * H_ + n;
            else     off = (size_t)m * Ncols + n;
            C[off] = v;
        }
    }
}

// ---------------------------------------------------------------- recurrent scan
// grid = 4 blocks (16 batch rows each), block = 512 thr (16 waves, 2 N-tiles each).
// h state (16x512 bf16) lives in LDS; W_hh streamed from L2 (512 KB, fully resident).
// xp is time-major f32 (already includes both biases).
__global__ __launch_bounds__(512) void rnn_scan(
    const float* __restrict__ xp, const bf16_t* __restrict__ whh,
    bf16_t* __restrict__ hout, float* __restrict__ hT)
{
    __shared__ bf16_t hbuf[16 * HPAD];
    const int lane = threadIdx.x & 31;
    const int wave = threadIdx.x >> 5;
    const int half = lane >> 4;
    const int nl   = lane & 15;
    const int b0   = blockIdx.x * 16;
    const int n_base = wave * 32;

    for (int i = threadIdx.x; i < 16 * HPAD; i += 512) hbuf[i] = (bf16_t)0.f;
    __syncthreads();

    for (int t = 0; t < T_; ++t) {
        // prefetch next timestep's xp slab (counter-free global_prefetch)
        if (t + 1 < T_) {
            const float* np = xp + ((size_t)(t + 1) * B_ + b0) * H_
                            + (size_t)(threadIdx.x >> 4) * H_ + (threadIdx.x & 15) * 16;
            __builtin_prefetch(np, 0, 1);
        }

        v8f acc[2];
        const float* xpt = xp + ((size_t)t * B_ + b0) * H_;
#pragma unroll
        for (int tt = 0; tt < 2; ++tt) {
            const int n = n_base + tt * 16 + nl;
#pragma unroll
            for (int r = 0; r < 8; ++r)
                acc[tt][r] = xpt[(size_t)(r + half * 8) * H_ + n];
        }

#pragma unroll 4
        for (int ks = 0; ks < 16; ++ks) {
            const bf16_t* ar = hbuf + nl * HPAD + ks * 32;
            v16bf af = pack16(*(const bf16x8*)(ar + half * 8),
                              *(const bf16x8*)(ar + 16 + half * 8));
#pragma unroll
            for (int tt = 0; tt < 2; ++tt) {
                const int n = n_base + tt * 16 + nl;
                const bf16_t* wp = whh + (size_t)n * H_ + ks * 32 + half * 8;
                v16bf bfm = pack16(*(const bf16x8*)wp, *(const bf16x8*)(wp + 16));
                acc[tt] = __builtin_amdgcn_wmma_f32_16x16x32_bf16(
                    false, af, false, bfm, (short)0, acc[tt], false, false);
            }
        }

        __syncthreads();   // everyone done reading h_{t-1} from LDS
#pragma unroll
        for (int tt = 0; tt < 2; ++tt) {
            const int n = n_base + tt * 16 + nl;
#pragma unroll
            for (int r = 0; r < 8; ++r) {
                const int m = r + half * 8;
                const float v = fast_tanh(acc[tt][r]);
                const bf16_t hv = (bf16_t)v;
                hbuf[m * HPAD + n] = hv;
                hout[((size_t)(b0 + m) * T_ + t) * H_ + n] = hv;
                if (t == T_ - 1) hT[(size_t)(b0 + m) * H_ + n] = v;
            }
        }
        __syncthreads();   // h_t visible before next step reads it
    }
}

// ---------------------------------------------------------------- launch
extern "C" void kernel_launch(void* const* d_in, const int* in_sizes, int n_in,
                              void* d_out, int out_size, void* d_ws, size_t ws_size,
                              hipStream_t stream) {
    const float* x     = (const float*)d_in[0];
    const float* w_ih0 = (const float*)d_in[1];
    const float* w_hh0 = (const float*)d_in[2];
    const float* b_ih0 = (const float*)d_in[3];
    const float* b_hh0 = (const float*)d_in[4];
    const float* w_ih1 = (const float*)d_in[5];
    const float* w_hh1 = (const float*)d_in[6];
    const float* b_ih1 = (const float*)d_in[7];
    const float* b_hh1 = (const float*)d_in[8];
    const float* w_out = (const float*)d_in[9];
    const float* b_out = (const float*)d_in[10];
    float* outf = (float*)d_out;

    // workspace carve-up
    char* ws = (char*)d_ws;
    float*  xp    = (float*)(ws);                                  // [T][B][H] f32, 128 MB (reused both layers)
    bf16_t* h1    = (bf16_t*)(ws + (size_t)134217728);             // [B][T][H] bf16, 64 MB
    bf16_t* h2    = (bf16_t*)(ws + (size_t)134217728 + 67108864);  // [B][T][H] bf16, 64 MB
    bf16_t* wih0b = (bf16_t*)(ws + (size_t)268435456);
    bf16_t* whh0b = wih0b + H_ * I_;
    bf16_t* wih1b = whh0b + H_ * H_;
    bf16_t* whh1b = wih1b + H_ * H_;
    bf16_t* woutb = whh1b + H_ * H_;
    float*  bsum0 = (float*)(woutb + O_ * H_);
    float*  bsum1 = bsum0 + H_;

    // 1) weight conversion f32 -> bf16 (row-major [N][K], matches B-fragment layout)
    cvt_f32_to_bf16<<<(H_ * I_ + 255) / 256, 256, 0, stream>>>(w_ih0, wih0b, H_ * I_);
    cvt_f32_to_bf16<<<(H_ * H_ + 255) / 256, 256, 0, stream>>>(w_hh0, whh0b, H_ * H_);
    cvt_f32_to_bf16<<<(H_ * H_ + 255) / 256, 256, 0, stream>>>(w_ih1, wih1b, H_ * H_);
    cvt_f32_to_bf16<<<(H_ * H_ + 255) / 256, 256, 0, stream>>>(w_hh1, whh1b, H_ * H_);
    cvt_f32_to_bf16<<<(O_ * H_ + 255) / 256, 256, 0, stream>>>(w_out, woutb, O_ * H_);
    bias_sum_k<<<1, H_, 0, stream>>>(b_ih0, b_hh0, bsum0);
    bias_sum_k<<<1, H_, 0, stream>>>(b_ih1, b_hh1, bsum1);

    const int M = B_ * T_;                 // 65536
    dim3 gProj(M / 128, H_ / 64);          // (512, 8)
    dim3 gOut(M / 128, 1);

    float* hT1 = outf + (size_t)B_ * T_ * O_;            // hidden[0]
    float* hT2 = hT1 + (size_t)B_ * H_;                  // hidden[1]

    // layer 0
    proj_gemm<true,  true ><<<gProj, 256, 0, stream>>>(x,  wih0b, bsum0, xp, I_, H_);
    rnn_scan<<<4, 512, 0, stream>>>(xp, whh0b, h1, hT1);
    // layer 1
    proj_gemm<false, true ><<<gProj, 256, 0, stream>>>(h1, wih1b, bsum1, xp, H_, H_);
    rnn_scan<<<4, 512, 0, stream>>>(xp, whh1b, h2, hT2);
    // output projection straight into d_out ([B][T][O] flat == m*O + n)
    proj_gemm<false, false><<<gOut, 256, 0, stream>>>(h2, woutb, b_out, outf, H_, O_);

    (void)in_sizes; (void)n_in; (void)out_size; (void)ws_size;
}
